// TransformerDecoder_80908593922488
// MI455X (gfx1250) — matrix operand order repaired and checked by reading, not compiled, not run on hardware
//
#include <hip/hip_runtime.h>

// ---------------------------------------------------------------------------
// Transformer decoder (6 layers) for MI455X / gfx1250.
// All matmuls run on v_wmma_f32_16x16x32_bf16 (bf16 in, fp32 accumulate).
// GEMM uses a double-buffered global->LDS async-DMA pipeline
// (global_load_async_to_lds_b128 + s_wait_asynccnt). fp32 kept for
// residual/LayerNorm state.
// ---------------------------------------------------------------------------

#define D_EMB 1024
#define N_HEAD 16
#define HEAD_D 64
#define F_FFN 4096
#define N_B 4
#define N_T 1024
#define N_S 1024
#define MROWS 4096          // B*T == B*S
#define N_LAYER 6

typedef __bf16 bf16;
typedef bf16  v16bf __attribute__((ext_vector_type(16)));
typedef float v8f   __attribute__((ext_vector_type(8)));
typedef int   td_v4i __attribute__((vector_size(16)));   // matches builtin param

#if defined(__has_builtin)
# if __has_builtin(__builtin_amdgcn_global_load_async_to_lds_b128) && \
     __has_builtin(__builtin_amdgcn_s_wait_asynccnt)
#  define TD_ASYNC_LDS 1
# endif
#endif
#ifndef TD_ASYNC_LDS
# define TD_ASYNC_LDS 0
#endif

#if TD_ASYNC_LDS
# define TD_NBUF 2
#else
# define TD_NBUF 1
#endif

// ---------------------------------------------------------------------------
// fp32 -> bf16 conversion (4-wide)
// ---------------------------------------------------------------------------
__global__ __launch_bounds__(256) void td_cvt_bf16(const float* __restrict__ s,
                                                   bf16* __restrict__ d, int n4) {
  int i = blockIdx.x * 256 + threadIdx.x;
  if (i >= n4) return;
  float4 f = ((const float4*)s)[i];
  union { bf16 e[4]; uint2 u; } o;
  o.e[0] = (bf16)f.x; o.e[1] = (bf16)f.y; o.e[2] = (bf16)f.z; o.e[3] = (bf16)f.w;
  ((uint2*)d)[i] = o.u;
}

// ---------------------------------------------------------------------------
// GEMM: C[M=4096, N] = act( (A[M,K](bf16) * W[N,K]^T(bf16) + bias) * scale )
// Block: 256 threads (8 wave32), C macro-tile 64x256, each wave 32x64 (2x4 WMMA).
// K-step 32; async double-buffered LDS staging when available.
// ACT: 0 = identity, 1 = tanh-approx GELU.
// ---------------------------------------------------------------------------
template <int ACT>
__global__ __launch_bounds__(256) void td_gemm_bf16(const bf16* __restrict__ A,
                                                    const bf16* __restrict__ W,
                                                    const float* __restrict__ bias,
                                                    float scale,
                                                    float* __restrict__ Cf,
                                                    bf16* __restrict__ Cb,
                                                    int N, int K) {
  __shared__ __attribute__((aligned(16))) bf16 As[TD_NBUF][64][40];
  __shared__ __attribute__((aligned(16))) bf16 Bs[TD_NBUF][256][40];

  const int tid  = threadIdx.x;
  const int lane = tid & 31, wvi = tid >> 5;
  const int wm = wvi & 1, wn = wvi >> 1;        // 2 (M) x 4 (N) wave grid
  const int hlf = lane >> 4, l15 = lane & 15;
  const size_t m0 = (size_t)blockIdx.y * 64;
  const size_t n0 = (size_t)blockIdx.x * 256;

  // cooperative staging: A 64x32 (1 x b128/thread), B 256x32 (4 x b128/thread)
  const int arow = tid >> 2, ak = (tid & 3) * 8;
  const int brow = tid;
  const bf16* aptr = A + (m0 + arow) * (size_t)K + ak;
  const bf16* bptr = W + (n0 + brow) * (size_t)K;

  v8f acc[2][4] = {};
  const int nIter = K / 32;

  auto compute_step = [&](int bsel) {
    v16bf afr[2], bfr[4];
    for (int mi = 0; mi < 2; ++mi) {
      // A-fragment (16x32): lane<16 -> K 0-7 & 16-23, lane>=16 -> K 8-15 & 24-31
      union { uint4 u[2]; v16bf v; } ua;
      const bf16* pA = &As[bsel][wm * 32 + mi * 16 + l15][hlf * 8];
      ua.u[0] = *(const uint4*)pA;
      ua.u[1] = *(const uint4*)(pA + 16);
      afr[mi] = ua.v;
    }
    for (int ni = 0; ni < 4; ++ni) {
      // B-fragment (32x16): lane = column, lane<16 -> K 0-7 & 16-23, lane>=16 -> K 8-15 & 24-31
      union { uint4 u[2]; v16bf v; } ub;
      const bf16* pB = &Bs[bsel][wn * 64 + ni * 16 + l15][hlf * 8];
      ub.u[0] = *(const uint4*)pB;
      ub.u[1] = *(const uint4*)(pB + 16);
      bfr[ni] = ub.v;
    }
    for (int mi = 0; mi < 2; ++mi)
      for (int ni = 0; ni < 4; ++ni)
        acc[mi][ni] = __builtin_amdgcn_wmma_f32_16x16x32_bf16(
            false, afr[mi], false, bfr[ni], (short)0, acc[mi][ni], false, false);
  };

#if TD_ASYNC_LDS
  // ---- double-buffered async global->LDS pipeline (ASYNCcnt-tracked DMA) ----
  auto issue = [&](int it) {
    const int bsel = it & 1;
    const bf16* ag = aptr + it * 32;
    const bf16* bg = bptr + it * 32;
    __builtin_amdgcn_global_load_async_to_lds_b128(
        (__attribute__((address_space(1))) td_v4i*)ag,
        (__attribute__((address_space(3))) td_v4i*)&As[bsel][arow][ak], 0, 0);
    for (int j = 0; j < 4; ++j)
      __builtin_amdgcn_global_load_async_to_lds_b128(
          (__attribute__((address_space(1))) td_v4i*)(bg + j * 8),
          (__attribute__((address_space(3))) td_v4i*)&Bs[bsel][brow][j * 8], 0, 0);
  };
  issue(0);
  for (int it = 0; it < nIter; ++it) {
    const int cur = it & 1;
    if (it + 1 < nIter) {
      issue(it + 1);                         // overlap next stage with compute
      __builtin_amdgcn_s_wait_asynccnt(5);   // in-order: stage `it` (5 ops) done
    } else {
      __builtin_amdgcn_s_wait_asynccnt(0);
    }
    __syncthreads();                         // all waves' stage-`it` DMA visible
    compute_step(cur);
    __syncthreads();                         // protect buffer before reuse
  }
#else
  // ---- fallback: synchronous regs->LDS staging with prefetch ----
  for (int it = 0; it < nIter; ++it) {
    const int k0 = it * 32;
    uint4 av  = *(const uint4*)(aptr + k0);
    uint4 bv0 = *(const uint4*)(bptr + k0);
    uint4 bv1 = *(const uint4*)(bptr + k0 + 8);
    uint4 bv2 = *(const uint4*)(bptr + k0 + 16);
    uint4 bv3 = *(const uint4*)(bptr + k0 + 24);
    if (k0 + 32 < K) {
      __builtin_prefetch(aptr + k0 + 32, 0, 1);   // global_prefetch_b8
      __builtin_prefetch(bptr + k0 + 32, 0, 1);
    }
    __syncthreads();
    *(uint4*)(&As[0][arow][ak])       = av;
    *(uint4*)(&Bs[0][brow][0])        = bv0;
    *(uint4*)(&Bs[0][brow][8])        = bv1;
    *(uint4*)(&Bs[0][brow][16])       = bv2;
    *(uint4*)(&Bs[0][brow][24])       = bv3;
    __syncthreads();
    compute_step(0);
  }
#endif

  // epilogue: C layout -> M = v + 8*hlf, Ncol = l15
  for (int mi = 0; mi < 2; ++mi)
    for (int ni = 0; ni < 4; ++ni) {
      size_t colg = n0 + wn * 64 + ni * 16 + l15;
      float bv = bias ? bias[colg] : 0.0f;
      for (int v = 0; v < 8; ++v) {
        size_t rowg = m0 + wm * 32 + mi * 16 + hlf * 8 + v;
        float x = (acc[mi][ni][v] + bv) * scale;
        if (ACT == 1) {
          float t = x + 0.044715f * x * x * x;
          x = 0.5f * x * (1.0f + tanhf(0.7978845608028654f * t));
        }
        if (Cf) Cf[rowg * N + colg] = x;
        if (Cb) Cb[rowg * N + colg] = (bf16)x;
      }
    }
}

// ---------------------------------------------------------------------------
// Flash attention: each wave handles 16 query rows of one (b,h); streams S in
// 32-column tiles. QK^T and P.V via WMMA; online softmax with 16-lane shuffles.
// Q is pre-scaled by HD^-0.5 at projection time.
// ---------------------------------------------------------------------------
template <int CAUSAL>
__global__ __launch_bounds__(256) void td_attn(const bf16* __restrict__ Q,
                                               const bf16* __restrict__ Km,
                                               const bf16* __restrict__ Vm,
                                               const float* __restrict__ kmask,
                                               bf16* __restrict__ O,
                                               int Skv) {
  __shared__ __attribute__((aligned(16))) bf16 vt[8][64][40];   // per-wave V^T tile
  __shared__ __attribute__((aligned(16))) bf16 pl[8][16][40];   // per-wave P tile

  const int tid  = threadIdx.x;
  const int lane = tid & 31, wv = tid >> 5;
  const int hlf = lane >> 4, l15 = lane & 15;

  const int bh = blockIdx.x / (N_T / 128);
  const int tb = blockIdx.x % (N_T / 128);
  const int b  = bh / N_HEAD, h = bh % N_HEAD;
  const int t0 = tb * 128 + wv * 16;
  const size_t qrow0  = (size_t)b * N_T + t0;
  const size_t kvrow0 = (size_t)b * Skv;

  // Q A-fragments for the two K-dim steps (HD=64 -> 2 x K32)
  v16bf qf[2];
  for (int kk = 0; kk < 2; ++kk) {
    union { uint4 u[2]; v16bf v; } uq;
    const bf16* p = Q + (qrow0 + l15) * D_EMB + h * HEAD_D + kk * 32 + hlf * 8;
    uq.u[0] = *(const uint4*)p;
    uq.u[1] = *(const uint4*)(p + 16);
    qf[kk] = uq.v;
  }

  v8f o[4] = {};
  float mrow[8], lrow[8];
  for (int v = 0; v < 8; ++v) { mrow[v] = -3.0e38f; lrow[v] = 0.0f; }

  const int smax = CAUSAL ? (t0 + 16) : Skv;
  for (int s0 = 0; s0 < smax; s0 += 32) {
    // stage V^T tile: lane loads V row (s0+lane), scatters into vt[d][lane]
    {
      const bf16* vp = Vm + (kvrow0 + s0 + lane) * D_EMB + h * HEAD_D;
      union { uint4 u[8]; bf16 e[64]; } uv;
      for (int j = 0; j < 8; ++j) uv.u[j] = *(const uint4*)(vp + j * 8);
      for (int d = 0; d < 64; ++d) vt[wv][d][lane] = uv.e[d];
    }

    // scores: S(16x32) = Q(16x64) * K^T(64x32), 2 N-tiles x 2 K-steps
    v8f sc[2] = {};
    for (int ni = 0; ni < 2; ++ni)
      for (int kk = 0; kk < 2; ++kk) {
        union { uint4 u[2]; v16bf v; } uk;
        const bf16* kp = Km + (kvrow0 + s0 + ni * 16 + l15) * D_EMB +
                         h * HEAD_D + kk * 32 + hlf * 8;
        uk.u[0] = *(const uint4*)kp;
        uk.u[1] = *(const uint4*)(kp + 16);
        sc[ni] = __builtin_amdgcn_wmma_f32_16x16x32_bf16(
            false, qf[kk], false, uk.v, (short)0, sc[ni], false, false);
      }

    // additive key-padding mask (per lane column)
    float addm[2];
    for (int ni = 0; ni < 2; ++ni) {
      int s = s0 + ni * 16 + l15;
      addm[ni] = (kmask[(size_t)b * Skv + s] > 0.0f) ? 0.0f : -3.0e38f;
    }

    // online softmax, per 8 rows owned by this lane half
    float pr0[8], pr1[8];
    for (int v = 0; v < 8; ++v) {
      int trow = t0 + hlf * 8 + v;
      float x0 = sc[0][v] + addm[0];
      float x1 = sc[1][v] + addm[1];
      if (CAUSAL) {
        if (s0 + l15 > trow)      x0 = -3.0e38f;
        if (s0 + 16 + l15 > trow) x1 = -3.0e38f;
      }
      float mx = fmaxf(x0, x1);
      for (int off = 8; off >= 1; off >>= 1) mx = fmaxf(mx, __shfl_xor(mx, off, 32));
      float mnew  = fmaxf(mrow[v], mx);
      float alpha = expf(mrow[v] - mnew);
      float p0 = expf(x0 - mnew);
      float p1 = expf(x1 - mnew);
      float ps = p0 + p1;
      for (int off = 8; off >= 1; off >>= 1) ps += __shfl_xor(ps, off, 32);
      lrow[v] = lrow[v] * alpha + ps;
      mrow[v] = mnew;
      for (int ni = 0; ni < 4; ++ni) o[ni][v] *= alpha;
      pr0[v] = p0; pr1[v] = p1;
    }

    // P (C layout) -> LDS -> A-fragment layout
    for (int v = 0; v < 8; ++v) {
      pl[wv][hlf * 8 + v][l15]      = (bf16)pr0[v];
      pl[wv][hlf * 8 + v][16 + l15] = (bf16)pr1[v];
    }
    union { uint4 u[2]; v16bf v; } up;
    {
      const bf16* pp = &pl[wv][l15][hlf * 8];
      up.u[0] = *(const uint4*)pp;
      up.u[1] = *(const uint4*)(pp + 16);
    }

    // O(16x64) += P(16x32) * V(32x64): 4 WMMA, B-fragments from transposed LDS
    for (int ni = 0; ni < 4; ++ni) {
      union { uint4 u[2]; v16bf v; } uvv;
      const bf16* pv = &vt[wv][ni * 16 + l15][hlf * 8];
      uvv.u[0] = *(const uint4*)pv;
      uvv.u[1] = *(const uint4*)(pv + 16);
      o[ni] = __builtin_amdgcn_wmma_f32_16x16x32_bf16(
          false, up.v, false, uvv.v, (short)0, o[ni], false, false);
    }
  }

  // normalize and store bf16 output
  for (int v = 0; v < 8; ++v) {
    float inv = lrow[v] > 0.0f ? 1.0f / lrow[v] : 0.0f;
    for (int ni = 0; ni < 4; ++ni)
      O[(qrow0 + hlf * 8 + v) * D_EMB + h * HEAD_D + ni * 16 + l15] =
          (bf16)(o[ni][v] * inv);
  }
}

// ---------------------------------------------------------------------------
// h = LayerNorm(R + X) * g + b, emitting fp32 state + bf16 GEMM input.
// One 256-thread block per row (D=1024 -> 4 elements/thread).
// ---------------------------------------------------------------------------
__global__ __launch_bounds__(256) void td_add_ln(const float* __restrict__ R,
                                                 const float* __restrict__ X,
                                                 const float* __restrict__ g,
                                                 const float* __restrict__ bta,
                                                 float* __restrict__ Hf,
                                                 bf16* __restrict__ Hb) {
  __shared__ float red[8];
  __shared__ float red2[8];
  const size_t row = blockIdx.x;
  const int tid = threadIdx.x, lane = tid & 31, wv = tid >> 5;
  const float* r = R + row * D_EMB;
  const float* x = X + row * D_EMB;

  float vals[4]; float s = 0.0f;
  for (int j = 0; j < 4; ++j) { int c = tid + j * 256; vals[j] = r[c] + x[c]; s += vals[j]; }
  for (int off = 16; off >= 1; off >>= 1) s += __shfl_xor(s, off, 32);
  if (lane == 0) red[wv] = s;
  __syncthreads();
  float tot = 0.0f;
  for (int i = 0; i < 8; ++i) tot += red[i];
  float mu = tot * (1.0f / D_EMB);

  float vs = 0.0f;
  for (int j = 0; j < 4; ++j) { float d2 = vals[j] - mu; vs += d2 * d2; }
  for (int off = 16; off >= 1; off >>= 1) vs += __shfl_xor(vs, off, 32);
  if (lane == 0) red2[wv] = vs;
  __syncthreads();
  float vtot = 0.0f;
  for (int i = 0; i < 8; ++i) vtot += red2[i];
  float rstd = rsqrtf(vtot * (1.0f / D_EMB) + 1e-5f);

  for (int j = 0; j < 4; ++j) {
    int c = tid + j * 256;
    float y = (vals[j] - mu) * rstd * g[c] + bta[c];
    Hf[row * D_EMB + c] = y;
    Hb[row * D_EMB + c] = (bf16)y;
  }
}

// ---------------------------------------------------------------------------
// Host orchestration
// ---------------------------------------------------------------------------
extern "C" void kernel_launch(void* const* d_in, const int* in_sizes, int n_in,
                              void* d_out, int out_size, void* d_ws, size_t ws_size,
                              hipStream_t stream) {
  (void)in_sizes; (void)n_in; (void)out_size; (void)ws_size;

  const float* hidden = (const float*)d_in[0];
  const float* enc    = (const float*)d_in[1];
  const float* amask  = (const float*)d_in[2];
  const float* emask  = (const float*)d_in[3];
  const float* attn_w[8];   // sa_wq, sa_wk, sa_wv, sa_wo, ca_wq, ca_wk, ca_wv, ca_wo
  const float* attn_b[8];
  for (int i = 0; i < 8; ++i) { attn_w[i] = (const float*)d_in[4 + i]; attn_b[i] = (const float*)d_in[12 + i]; }
  const float* ln_g[3] = { (const float*)d_in[20], (const float*)d_in[22], (const float*)d_in[24] };
  const float* ln_b[3] = { (const float*)d_in[21], (const float*)d_in[23], (const float*)d_in[25] };
  const float* fc1_w = (const float*)d_in[26];
  const float* fc1_b = (const float*)d_in[27];
  const float* fc2_w = (const float*)d_in[28];
  const float* fc2_b = (const float*)d_in[29];

  // ---- workspace carve (~145 MB) ----
  char* p = (char*)d_ws;
  auto carve = [&](size_t bytes) -> void* {
    void* r = (void*)p; p += (bytes + 255) & ~(size_t)255; return r;
  };
  float* hF    = (float*)carve((size_t)MROWS * D_EMB * 4);
  bf16*  hB    = (bf16*) carve((size_t)MROWS * D_EMB * 2);
  bf16*  encB  = (bf16*) carve((size_t)MROWS * D_EMB * 2);
  bf16*  qB    = (bf16*) carve((size_t)MROWS * D_EMB * 2);
  bf16*  kB    = (bf16*) carve((size_t)MROWS * D_EMB * 2);
  bf16*  vB    = (bf16*) carve((size_t)MROWS * D_EMB * 2);
  bf16*  oB    = (bf16*) carve((size_t)MROWS * D_EMB * 2);
  float* aF    = (float*)carve((size_t)MROWS * D_EMB * 4);
  bf16*  fB    = (bf16*) carve((size_t)MROWS * F_FFN * 2);
  bf16*  wAtt[8];
  for (int i = 0; i < 8; ++i) wAtt[i] = (bf16*)carve((size_t)D_EMB * D_EMB * 2);
  bf16*  wFc1  = (bf16*)carve((size_t)F_FFN * D_EMB * 2);
  bf16*  wFc2  = (bf16*)carve((size_t)D_EMB * F_FFN * 2);

  auto cvt = [&](const float* src, bf16* dst, size_t n) {
    int n4 = (int)(n / 4);
    td_cvt_bf16<<<dim3((n4 + 255) / 256), dim3(256), 0, stream>>>(src, dst, n4);
  };

  const dim3 blk(256);
  const dim3 gD(D_EMB / 256, MROWS / 64);    // GEMM grid for N=1024
  const dim3 gF(F_FFN / 256, MROWS / 64);    // GEMM grid for N=4096
  const dim3 gA(N_B * N_HEAD * (N_T / 128)); // attention grid (512 blocks)
  const float qscale = 0.125f;               // HD^-0.5

  // init state
  (void)hipMemcpyAsync(hF, hidden, (size_t)MROWS * D_EMB * 4, hipMemcpyDeviceToDevice, stream);
  cvt(hidden, hB, (size_t)MROWS * D_EMB);
  cvt(enc, encB, (size_t)MROWS * D_EMB);

  for (int i = 0; i < N_LAYER; ++i) {
    const size_t wOff = (size_t)i * D_EMB * D_EMB;
    const size_t bOff = (size_t)i * D_EMB;
    for (int j = 0; j < 8; ++j) cvt(attn_w[j] + wOff, wAtt[j], (size_t)D_EMB * D_EMB);
    cvt(fc1_w + (size_t)i * F_FFN * D_EMB, wFc1, (size_t)F_FFN * D_EMB);
    cvt(fc2_w + (size_t)i * D_EMB * F_FFN, wFc2, (size_t)D_EMB * F_FFN);

    // ---- self-attention ----
    td_gemm_bf16<0><<<gD, blk, 0, stream>>>(hB, wAtt[0], attn_b[0] + bOff, qscale,
                                            (float*)nullptr, qB, D_EMB, D_EMB);
    td_gemm_bf16<0><<<gD, blk, 0, stream>>>(hB, wAtt[1], attn_b[1] + bOff, 1.0f,
                                            (float*)nullptr, kB, D_EMB, D_EMB);
    td_gemm_bf16<0><<<gD, blk, 0, stream>>>(hB, wAtt[2], attn_b[2] + bOff, 1.0f,
                                            (float*)nullptr, vB, D_EMB, D_EMB);
    td_attn<1><<<gA, blk, 0, stream>>>(qB, kB, vB, amask, oB, N_T);
    td_gemm_bf16<0><<<gD, blk, 0, stream>>>(oB, wAtt[3], attn_b[3] + bOff, 1.0f,
                                            aF, (bf16*)nullptr, D_EMB, D_EMB);
    td_add_ln<<<dim3(MROWS), blk, 0, stream>>>(hF, aF, ln_g[0] + bOff, ln_b[0] + bOff, hF, hB);

    // ---- cross-attention ----
    td_gemm_bf16<0><<<gD, blk, 0, stream>>>(hB, wAtt[4], attn_b[4] + bOff, qscale,
                                            (float*)nullptr, qB, D_EMB, D_EMB);
    td_gemm_bf16<0><<<gD, blk, 0, stream>>>(encB, wAtt[5], attn_b[5] + bOff, 1.0f,
                                            (float*)nullptr, kB, D_EMB, D_EMB);
    td_gemm_bf16<0><<<gD, blk, 0, stream>>>(encB, wAtt[6], attn_b[6] + bOff, 1.0f,
                                            (float*)nullptr, vB, D_EMB, D_EMB);
    td_attn<0><<<gA, blk, 0, stream>>>(qB, kB, vB, emask, oB, N_S);
    td_gemm_bf16<0><<<gD, blk, 0, stream>>>(oB, wAtt[7], attn_b[7] + bOff, 1.0f,
                                            aF, (bf16*)nullptr, D_EMB, D_EMB);
    td_add_ln<<<dim3(MROWS), blk, 0, stream>>>(hF, aF, ln_g[1] + bOff, ln_b[1] + bOff, hF, hB);

    // ---- FFN ----
    td_gemm_bf16<1><<<gF, blk, 0, stream>>>(hB, wFc1, fc1_b + (size_t)i * F_FFN, 1.0f,
                                            (float*)nullptr, fB, F_FFN, D_EMB);
    td_gemm_bf16<0><<<gD, blk, 0, stream>>>(fB, wFc2, fc2_b + bOff, 1.0f,
                                            aF, (bf16*)nullptr, D_EMB, F_FFN);
    td_add_ln<<<dim3(MROWS), blk, 0, stream>>>(hF, aF, ln_g[2] + bOff, ln_b[2] + bOff, hF, hB);
  }

  (void)hipMemcpyAsync(d_out, hF, (size_t)MROWS * D_EMB * 4, hipMemcpyDeviceToDevice, stream);
}
